// GetAttnScores_41798621724782
// MI455X (gfx1250) — matrix operand order repaired and checked by reading, not compiled, not run on hardware
//
#include <hip/hip_runtime.h>

typedef __attribute__((ext_vector_type(16))) _Float16 v16h;
typedef __attribute__((ext_vector_type(4)))  _Float16 v4h;
typedef __attribute__((ext_vector_type(8)))  float    v8f;

#define BATCH 2
#define HQ    32
#define HKV   8
#define SEQ   2048
#define DIM   128
#define SW    (SEQ / 32)  // mask bit-words per row
// DIM + 8 halfs of padding -> 272B LDS row stride: 68 dwords, so lane n sits on
// bank 4n for its b128 read -> conflict-free ds_load_b128 fragment fetches.
#define LDS_STRIDE 136

union FragH {
  v16h v;
  uint4 q[2];
};

__device__ __forceinline__ v8f wmma_f16(const v16h& a, const v16h& b, v8f c) {
  // 8-arg form: (neg_a, A, neg_b, B, c_mod, C, reuse_a, reuse_b)
  return __builtin_amdgcn_wmma_f32_16x16x32_f16(false, a, false, b, (short)0, c,
                                                false, false);
}

// ---------------------------------------------------------------------------
// Pass 0a: f32 -> f16 conversion of Q and K into workspace.
// ---------------------------------------------------------------------------
__global__ __launch_bounds__(256) void cvt_kernel(const float* __restrict__ qf,
                                                  const float* __restrict__ kf,
                                                  _Float16* __restrict__ qh,
                                                  _Float16* __restrict__ kh) {
  const size_t NQ = (size_t)BATCH * HQ * SEQ * DIM;
  const size_t NK = (size_t)BATCH * HKV * SEQ * DIM;
  size_t i = ((size_t)blockIdx.x * 256 + threadIdx.x) * 4;
  if (i < NQ) {
    float4 v = *(const float4*)(qf + i);
    v4h hx = {(_Float16)v.x, (_Float16)v.y, (_Float16)v.z, (_Float16)v.w};
    *(v4h*)(qh + i) = hx;
  } else if (i < NQ + NK) {
    size_t j = i - NQ;
    float4 v = *(const float4*)(kf + j);
    v4h hx = {(_Float16)v.x, (_Float16)v.y, (_Float16)v.z, (_Float16)v.w};
    *(v4h*)(kh + j) = hx;
  }
}

// ---------------------------------------------------------------------------
// Pass 0b: pack the byte bool mask into bits (1 u32 = 32 columns). Mask is
// head-invariant, so both GEMM passes can then read it with wide loads.
// ---------------------------------------------------------------------------
__global__ __launch_bounds__(256) void maskpack_kernel(
    const unsigned char* __restrict__ mask, unsigned int* __restrict__ mbits) {
  const size_t NW = (size_t)BATCH * SEQ * SW;
  size_t w = (size_t)blockIdx.x * 256 + threadIdx.x;
  if (w >= NW) return;
  const unsigned char* src = mask + w * 32;
  uint4 a = *(const uint4*)(src);
  uint4 c = *(const uint4*)(src + 16);
  unsigned int dw[8] = {a.x, a.y, a.z, a.w, c.x, c.y, c.z, c.w};
  unsigned int bits = 0;
#pragma unroll
  for (int d = 0; d < 8; ++d) {
    bits |= (dw[d] & 1u) << (4 * d);
    bits |= ((dw[d] >> 8) & 1u) << (4 * d + 1);
    bits |= ((dw[d] >> 16) & 1u) << (4 * d + 2);
    bits |= ((dw[d] >> 24) & 1u) << (4 * d + 3);
  }
  mbits[w] = bits;
}

// ---------------------------------------------------------------------------
// Pass 1: online-softmax row statistics (max m, sum-exp l) per (b, h, q-row).
// Block = (b, h, 128 q-rows); 8 waves x 16 rows; streams all 2048 k in
// 16-wide WMMA tiles, K staged 64 rows at a time through LDS.
// ---------------------------------------------------------------------------
__global__ __launch_bounds__(256) void attn_stats_kernel(
    const _Float16* __restrict__ qh, const _Float16* __restrict__ kh,
    const unsigned int* __restrict__ mbits, const float* __restrict__ scale_p,
    float2* __restrict__ stats) {
  __shared__ __align__(16) _Float16 ktile[64 * LDS_STRIDE];
  const float scale = *scale_p;
  const int blk = blockIdx.x;
  const int qt = blk & 15;
  const int h  = (blk >> 4) & 31;
  const int b  = blk >> 9;
  const int wave = threadIdx.x >> 5;
  const int lane = threadIdx.x & 31;
  const int m  = lane & 15;   // A: row M; B/mask: col N
  const int hi = lane >> 4;   // half-wave select
  const int qr0 = qt * 128 + wave * 16;

  // A fragments (Q, 16 rows x full D=128), resident for the whole k loop.
  // ISA layout: lanes<16 hold K 0-7 & 16-23 (VGPR 0-3 / 4-7), lanes>=16 hold
  // K 8-15 & 24-31 -> two b128 loads per 32-wide K chunk.
  const _Float16* qrow = qh + ((size_t)(b * HQ + h) * SEQ + qr0 + m) * DIM;
  FragH A[4];
#pragma unroll
  for (int f = 0; f < 4; ++f) {
    A[f].q[0] = *(const uint4*)(qrow + 32 * f + 8 * hi);
    A[f].q[1] = *(const uint4*)(qrow + 32 * f + 16 + 8 * hi);
  }
  const _Float16* kbase = kh + (size_t)(b * HKV + (h >> 2)) * SEQ * DIM;
  const unsigned int* mb = mbits + (size_t)b * SEQ * SW;

  float mrun[8], lrun[8];
#pragma unroll
  for (int j = 0; j < 8; ++j) { mrun[j] = -__builtin_inff(); lrun[j] = 0.f; }

  for (int kc = 0; kc < SEQ; kc += 64) {
    __syncthreads();
#pragma unroll
    for (int c = 0; c < 4; ++c) {  // stage 64 K rows (16 KB) cooperatively
      int chunk = (int)threadIdx.x + 256 * c;
      int row = chunk >> 4;
      int off = (chunk & 15) * 8;
      *(uint4*)(ktile + row * LDS_STRIDE + off) =
          *(const uint4*)(kbase + (size_t)(kc + row) * DIM + off);
    }
    __syncthreads();
    // mask bits for this 64-column chunk: one 8-byte load per row
    unsigned int mw0[8], mw1[8];
#pragma unroll
    for (int j = 0; j < 8; ++j) {
      uint2 t = *(const uint2*)&mb[(size_t)(qr0 + j + 8 * hi) * SW + (kc >> 5)];
      mw0[j] = t.x;
      mw1[j] = t.y;
    }
#pragma unroll
    for (int tt = 0; tt < 4; ++tt) {
      v8f acc = {};
#pragma unroll
      for (int f = 0; f < 4; ++f) {
        // B layout: lane = N (key index), lanes<16 hold K 0-7 & 16-23, lanes>=16 K 8-15 & 24-31
        const _Float16* bp = ktile + (tt * 16 + m) * LDS_STRIDE + 32 * f + 8 * hi;
        FragH Bf;
        Bf.q[0] = *(const uint4*)(bp);
        Bf.q[1] = *(const uint4*)(bp + 16);
        acc = wmma_f16(A[f].v, Bf.v, acc);
      }
#pragma unroll
      for (int j = 0; j < 8; ++j) {  // C layout: VGPR j -> row j + 8*hi
        float s = acc[j] * scale;
        const unsigned int word = (tt & 2) ? mw1[j] : mw0[j];
        if ((word >> (((tt & 1) << 4) + m)) & 1u) s = -__builtin_inff();
        float tmax = s;  // reduce over the 16-lane N group (xor masks <= 8)
        tmax = fmaxf(tmax, __shfl_xor(tmax, 1, 32));
        tmax = fmaxf(tmax, __shfl_xor(tmax, 2, 32));
        tmax = fmaxf(tmax, __shfl_xor(tmax, 4, 32));
        tmax = fmaxf(tmax, __shfl_xor(tmax, 8, 32));
        const float newm = fmaxf(mrun[j], tmax);
        float p = (newm == -__builtin_inff()) ? 0.f : __expf(s - newm);
        float corr = (mrun[j] == -__builtin_inff())
                         ? ((newm == -__builtin_inff()) ? 1.f : 0.f)
                         : __expf(mrun[j] - newm);
        float ps = p;
        ps += __shfl_xor(ps, 1, 32);
        ps += __shfl_xor(ps, 2, 32);
        ps += __shfl_xor(ps, 4, 32);
        ps += __shfl_xor(ps, 8, 32);
        lrun[j] = lrun[j] * corr + ps;
        mrun[j] = newm;
      }
    }
  }
  if (m == 0) {
#pragma unroll
    for (int j = 0; j < 8; ++j) {
      const int row = qr0 + j + 8 * hi;
      stats[(size_t)(b * HQ + h) * SEQ + row] = make_float2(mrun[j], lrun[j]);
    }
  }
}

// ---------------------------------------------------------------------------
// Pass 2: recompute scores, apply exp(s-m)/l, accumulate over the 32 heads in
// registers, write out once. Block = (b, 128 q-rows, 128 k-cols); K tile
// (32 KB) stays LDS-resident while the 4 heads of an Hkv group are processed.
// Mask bits for the whole block tile are loaded ONCE (head-invariant).
// ---------------------------------------------------------------------------
__global__ __launch_bounds__(256) void attn_emit_kernel(
    const _Float16* __restrict__ qh, const _Float16* __restrict__ kh,
    const unsigned int* __restrict__ mbits, const float* __restrict__ scale_p,
    const float2* __restrict__ stats, float* __restrict__ out) {
  __shared__ __align__(16) _Float16 ktile[128 * LDS_STRIDE];
  const float scale = *scale_p;
  const int blk = blockIdx.x;
  const int ktB = blk & 15;
  const int qt  = (blk >> 4) & 15;
  const int b   = blk >> 8;
  const int wave = threadIdx.x >> 5;
  const int lane = threadIdx.x & 31;
  const int m  = lane & 15;
  const int hi = lane >> 4;
  const int qr0 = qt * 128 + wave * 16;
  const int kb0 = ktB * 128;

  // Head-invariant mask bits for this 128x128 tile: one b128 load per row.
  const unsigned int* mb = mbits + (size_t)b * SEQ * SW;
  unsigned int mw[8][4];
#pragma unroll
  for (int j = 0; j < 8; ++j) {
    uint4 t = *(const uint4*)&mb[(size_t)(qr0 + j + 8 * hi) * SW + (kb0 >> 5)];
    mw[j][0] = t.x; mw[j][1] = t.y; mw[j][2] = t.z; mw[j][3] = t.w;
  }

  v8f zero = {};
  v8f oacc[8];
#pragma unroll
  for (int t = 0; t < 8; ++t) oacc[t] = zero;

  for (int h = 0; h < HQ; ++h) {
    if ((h & 3) == 0) {  // new Hkv group: stage its 128x128 K tile (32 KB)
      __syncthreads();
      const _Float16* kbase =
          kh + ((size_t)(b * HKV + (h >> 2)) * SEQ + kb0) * DIM;
#pragma unroll
      for (int c = 0; c < 8; ++c) {
        int chunk = (int)threadIdx.x + 256 * c;
        int row = chunk >> 4;
        int off = (chunk & 15) * 8;
        *(uint4*)(ktile + row * LDS_STRIDE + off) =
            *(const uint4*)(kbase + (size_t)row * DIM + off);
      }
      __syncthreads();
    }
    const _Float16* qrow = qh + ((size_t)(b * HQ + h) * SEQ + qr0 + m) * DIM;
    FragH A[4];
#pragma unroll
    for (int f = 0; f < 4; ++f) {
      A[f].q[0] = *(const uint4*)(qrow + 32 * f + 8 * hi);
      A[f].q[1] = *(const uint4*)(qrow + 32 * f + 16 + 8 * hi);
    }
    float mh[8], il[8];
#pragma unroll
    for (int j = 0; j < 8; ++j) {
      float2 st = stats[(size_t)(b * HQ + h) * SEQ + qr0 + j + 8 * hi];
      mh[j] = st.x;
      il[j] = st.y > 0.f ? 1.f / st.y : 0.f;
    }
#pragma unroll
    for (int kt = 0; kt < 8; ++kt) {
      v8f acc = {};
#pragma unroll
      for (int f = 0; f < 4; ++f) {
        const _Float16* bp = ktile + (kt * 16 + m) * LDS_STRIDE + 32 * f + 8 * hi;
        FragH Bf;
        Bf.q[0] = *(const uint4*)(bp);
        Bf.q[1] = *(const uint4*)(bp + 16);
        acc = wmma_f16(A[f].v, Bf.v, acc);
      }
#pragma unroll
      for (int j = 0; j < 8; ++j) {
        float s = acc[j] * scale;
        const unsigned int bit = (mw[j][kt >> 1] >> (((kt & 1) << 4) + m)) & 1u;
        float p = (bit || mh[j] == -__builtin_inff())
                      ? 0.f
                      : __expf(s - mh[j]) * il[j];
        oacc[kt][j] += p;
      }
    }
  }
#pragma unroll
  for (int kt = 0; kt < 8; ++kt) {
#pragma unroll
    for (int j = 0; j < 8; ++j) {
      out[((size_t)b * SEQ + qr0 + j + 8 * hi) * SEQ + kb0 + kt * 16 + m] =
          oacc[kt][j];
    }
  }
}

// ---------------------------------------------------------------------------
extern "C" void kernel_launch(void* const* d_in, const int* in_sizes, int n_in,
                              void* d_out, int out_size, void* d_ws,
                              size_t ws_size, hipStream_t stream) {
  const float* q = (const float*)d_in[0];
  const float* k = (const float*)d_in[1];
  const unsigned char* mask = (const unsigned char*)d_in[2];
  // d_in[3] = num_attn_head_per_group (structure hardcoded: 32/8 = 4)
  const float* scale = (const float*)d_in[4];

  // Workspace: Qh (33.6 MB) | Kh (8.4 MB) | stats (1 MB) | mask bits (1 MB)
  _Float16* qh = (_Float16*)d_ws;
  _Float16* kh = qh + (size_t)BATCH * HQ * SEQ * DIM;
  float2* stats = (float2*)(kh + (size_t)BATCH * HKV * SEQ * DIM);
  unsigned int* mbits = (unsigned int*)(stats + (size_t)BATCH * HQ * SEQ);
  float* out = (float*)d_out;

  const size_t total4 = ((size_t)BATCH * (HQ + HKV) * SEQ * DIM) / 4;
  cvt_kernel<<<dim3((unsigned)((total4 + 255) / 256)), dim3(256), 0, stream>>>(
      q, k, qh, kh);
  const size_t nwords = (size_t)BATCH * SEQ * SW;
  maskpack_kernel<<<dim3((unsigned)((nwords + 255) / 256)), dim3(256), 0,
                    stream>>>(mask, mbits);
  attn_stats_kernel<<<dim3(BATCH * HQ * (SEQ / 128)), dim3(256), 0, stream>>>(
      qh, kh, mbits, scale, stats);
  attn_emit_kernel<<<dim3(BATCH * 16 * 16), dim3(256), 0, stream>>>(
      qh, kh, mbits, scale, stats, out);
}
